// IdealDPM_32349693673941
// MI455X (gfx1250) — matrix operand order repaired and checked by reading, not compiled, not run on hardware
//
#include <hip/hip_runtime.h>
#include <math.h>

#define Bv   8
#define Tv   2048
#define DINv 256
#define Ev   256
#define Hv   4
#define Pv   8
#define HDv  64
#define Mv   (Bv * Tv)   // 16384 rows

typedef __attribute__((ext_vector_type(16))) __bf16 v16bf;
typedef __attribute__((ext_vector_type(8)))  float  v8f;

union BF16x16 { v16bf v; unsigned short s[16]; uint4 q[2]; };
union F32x8   { v8f  v; float f[8]; };

__device__ __forceinline__ unsigned short f2bf(float f) {
  return __builtin_bit_cast(unsigned short, static_cast<__bf16>(f));
}

// ---------------- WMMA fragment loaders (wave32 layouts per CDNA5 ISA 7.12.2) --
// A (16x32 MxK) bf16, row-major source [*, lda]
__device__ __forceinline__ v16bf load_a_bf16(const unsigned short* A, int lda,
                                             int m0, int k0, int lane) {
  const int row = m0 + (lane & 15);
  const int kb  = k0 + ((lane >> 4) << 3);
  const unsigned short* p = A + (size_t)row * lda + kb;
  BF16x16 r;
  r.q[0] = *reinterpret_cast<const uint4*>(p);
  r.q[1] = *reinterpret_cast<const uint4*>(p + 16);
  return r.v;
}

// B (32x16 KxN) where B[k][n] = W[n0+n][k0+k]; W row-major [n][ldk] bf16
__device__ __forceinline__ v16bf load_bT_bf16(const unsigned short* W, int ldk,
                                              int n0, int k0, int lane) {
  const int col = n0 + (lane & 15);
  const int kb  = k0 + ((lane >> 4) << 3);
  const unsigned short* p = W + (size_t)col * ldk + kb;
  BF16x16 r;
  r.q[0] = *reinterpret_cast<const uint4*>(p);
  r.q[1] = *reinterpret_cast<const uint4*>(p + 16);
  return r.v;
}

// A fragment from a 16x32 row-major bf16 tile in LDS (P-tile restage)
__device__ __forceinline__ v16bf load_a_lds(const unsigned short* p_lds, int lane) {
  const int row = lane & 15;
  const int kb  = (lane >> 4) << 3;
  BF16x16 r;
#pragma unroll
  for (int e = 0; e < 8; ++e) {
    r.s[e]     = p_lds[row * 32 + kb + e];
    r.s[8 + e] = p_lds[row * 32 + kb + 16 + e];
  }
  return r.v;
}

__device__ __forceinline__ v8f wmma_bf16(v16bf a, v16bf b, v8f c) {
  return __builtin_amdgcn_wmma_f32_16x16x32_bf16(false, a, false, b,
                                                 (short)0, c, false, false);
}

// half-wave (16-lane group) reductions
__device__ __forceinline__ float hmax16(float v) {
#pragma unroll
  for (int m = 1; m < 16; m <<= 1) v = fmaxf(v, __shfl_xor(v, m, 16));
  return v;
}
__device__ __forceinline__ float hsum16(float v) {
#pragma unroll
  for (int m = 1; m < 16; m <<= 1) v += __shfl_xor(v, m, 16);
  return v;
}

// ---- 32x64 GEMM core: C[2g+i] = tile (m0+16i, n0+16g), B loads pipelined -----
template <int K>
__device__ __forceinline__ void gemm32x64(const unsigned short* __restrict__ A, int lda,
                                          const unsigned short* __restrict__ W,
                                          int m0, int n0, int lane, v8f c[8]) {
  v16bf bcur = load_bT_bf16(W, K, n0, 0, lane);
#pragma unroll
  for (int k0 = 0; k0 < K; k0 += 32) {
    const v16bf a0 = load_a_bf16(A, lda, m0,      k0, lane);
    const v16bf a1 = load_a_bf16(A, lda, m0 + 16, k0, lane);
#pragma unroll
    for (int g = 0; g < 4; ++g) {
      v16bf bnext = bcur;
      const int gn = (g + 1) & 3;
      const int kn = (g == 3) ? (k0 + 32) : k0;
      if (kn < K) bnext = load_bT_bf16(W, K, n0 + gn * 16, kn, lane);  // in flight
      c[2 * g]     = wmma_bf16(a0, bcur, c[2 * g]);                    // during these
      c[2 * g + 1] = wmma_bf16(a1, bcur, c[2 * g + 1]);
      bcur = bnext;
    }
  }
}

// ============ P0: f32 -> bf16 bulk convert (4 elems/thread) ===================
__global__ void k_cvt(const float* __restrict__ src, unsigned short* __restrict__ dst,
                      int n4) {
  const int i = blockIdx.x * blockDim.x + threadIdx.x;
  if (i >= n4) return;
  const float4 v = reinterpret_cast<const float4*>(src)[i];
  uint2 o;
  o.x = (unsigned)f2bf(v.x) | ((unsigned)f2bf(v.y) << 16);
  o.y = (unsigned)f2bf(v.z) | ((unsigned)f2bf(v.w) << 16);
  reinterpret_cast<uint2*>(dst)[i] = o;
}

// ============ P1: fold w_o/b_o into w_bd/b_bd (linear chain collapse) =========
__global__ void k_fold(const float* __restrict__ w_o, const float* __restrict__ b_o,
                       const float* __restrict__ w_bd, const float* __restrict__ b_bd,
                       float* __restrict__ weff) {
  const int kk = threadIdx.x;   // 256
  float acc = 0.f;
  for (int n = 0; n < Ev; ++n) acc += w_o[(size_t)n * Ev + kk] * w_bd[n];
  weff[kk] = acc;
  if (kk == 0) {
    float s = b_bd[0];
    for (int n = 0; n < Ev; ++n) s += b_o[n] * w_bd[n];
    weff[Ev] = s;
  }
}

// ============ K1: h = x @ w_in^T + b_in -> h_f32, h_bf16 (32x64 tile/wave) ====
__global__ void k_hproj(const unsigned short* __restrict__ x16,
                        const unsigned short* __restrict__ w16,
                        const float* __restrict__ bias,
                        float* __restrict__ h32, unsigned short* __restrict__ h16) {
  const int lane = threadIdx.x & 31;
  const int m0 = blockIdx.x * 32;
  const int n0 = blockIdx.y * 64;
  v8f c[8] = {};
  gemm32x64<DINv>(x16, DINv, w16, m0, n0, lane, c);
#pragma unroll
  for (int g = 0; g < 4; ++g) {
    const int col = n0 + g * 16 + (lane & 15);
    const float bn = bias[col];
#pragma unroll
    for (int i = 0; i < 2; ++i) {
      F32x8 C; C.v = c[2 * g + i];
#pragma unroll
      for (int r = 0; r < 8; ++r) {
        const int row = m0 + i * 16 + r + ((lane >> 4) << 3);
        const float v = C.f[r] + bn;
        h32[(size_t)row * Ev + col] = v;
        h16[(size_t)row * Ev + col] = f2bf(v);
      }
    }
  }
}

// ============ K2: q/k/v projections (z = which), V stored transposed ==========
__global__ void k_qkv(const unsigned short* __restrict__ h16,
                      const unsigned short* __restrict__ wq16,
                      const unsigned short* __restrict__ wk16,
                      const unsigned short* __restrict__ wv16,
                      const float* __restrict__ bq, const float* __restrict__ bk,
                      const float* __restrict__ bv,
                      unsigned short* __restrict__ q, unsigned short* __restrict__ k,
                      unsigned short* __restrict__ vT) {
  const int lane  = threadIdx.x & 31;
  const int which = blockIdx.z;
  const unsigned short* w = (which == 0) ? wq16 : (which == 1) ? wk16 : wv16;
  const float* bb         = (which == 0) ? bq   : (which == 1) ? bk   : bv;
  const int m0 = blockIdx.x * 32;
  const int n0 = blockIdx.y * 64;       // == head * 64
  v8f c[8] = {};
  gemm32x64<Ev>(h16, Ev, w, m0, n0, lane, c);
  const int head = blockIdx.y;
#pragma unroll
  for (int g = 0; g < 4; ++g) {
    const int col = n0 + g * 16 + (lane & 15);
    const int d   = g * 16 + (lane & 15);
    const float bn = bb[col];
#pragma unroll
    for (int i = 0; i < 2; ++i) {
      F32x8 C; C.v = c[2 * g + i];
#pragma unroll
      for (int r = 0; r < 8; ++r) {
        const int row = m0 + i * 16 + r + ((lane >> 4) << 3);
        const int bi = row >> 11, t = row & (Tv - 1);
        const int bh = bi * Hv + head;
        const unsigned short v = f2bf(C.f[r] + bn);
        if (which == 2) vT[((size_t)bh * HDv + d) * Tv + t] = v;
        else {
          unsigned short* dst = (which == 0) ? q : k;
          dst[((size_t)bh * Tv + t) * HDv + d] = v;
        }
      }
    }
  }
}

// ============ K3: flash attention + fused boundary-score reduction ============
__global__ void k_attn(const unsigned short* __restrict__ q,
                       const unsigned short* __restrict__ kk,
                       const unsigned short* __restrict__ vT,
                       const float* __restrict__ weff,
                       float* __restrict__ part) {
  __shared__ unsigned short p_lds[16 * 32];           // P-tile restage, 1 KB
  const int lane = threadIdx.x & 31;
  const int bh = blockIdx.x;                          // [0, B*H)
  const int q0 = blockIdx.y * 16;
  const unsigned short* qb = q  + (size_t)bh * Tv * HDv;
  const unsigned short* kb = kk + (size_t)bh * Tv * HDv;
  const unsigned short* vb = vT + (size_t)bh * HDv * Tv;
  const float scale = 0.125f;                         // 1/sqrt(64)

  const v16bf aQ0 = load_a_bf16(qb, HDv, q0, 0,  lane);
  const v16bf aQ1 = load_a_bf16(qb, HDv, q0, 32, lane);

  float mrun[8], lrun[8];
#pragma unroll
  for (int r = 0; r < 8; ++r) { mrun[r] = -3.4e38f; lrun[r] = 0.f; }
  v8f a0 = {}, a1 = {}, a2 = {}, a3 = {};

  // software-pipelined K-tile fragments
  v16bf bK0 = load_bT_bf16(kb, HDv, 0,  0,  lane);
  v16bf bK1 = load_bT_bf16(kb, HDv, 0,  32, lane);
  v16bf bK2 = load_bT_bf16(kb, HDv, 16, 0,  lane);
  v16bf bK3 = load_bT_bf16(kb, HDv, 16, 32, lane);

  for (int kt = 0; kt < Tv; kt += 32) {
    v8f s0 = {}, s1 = {};
    s0 = wmma_bf16(aQ0, bK0, s0);
    s0 = wmma_bf16(aQ1, bK1, s0);
    s1 = wmma_bf16(aQ0, bK2, s1);
    s1 = wmma_bf16(aQ1, bK3, s1);

    // issue V-tile and next K-tile loads NOW; they overlap the softmax VALU work
    const v16bf bV0 = load_bT_bf16(vb, Tv, 0,  kt, lane);
    const v16bf bV1 = load_bT_bf16(vb, Tv, 16, kt, lane);
    const v16bf bV2 = load_bT_bf16(vb, Tv, 32, kt, lane);
    const v16bf bV3 = load_bT_bf16(vb, Tv, 48, kt, lane);
    if (kt + 32 < Tv) {
      bK0 = load_bT_bf16(kb, HDv, kt + 32, 0,  lane);
      bK1 = load_bT_bf16(kb, HDv, kt + 32, 32, lane);
      bK2 = load_bT_bf16(kb, HDv, kt + 48, 0,  lane);
      bK3 = load_bT_bf16(kb, HDv, kt + 48, 32, lane);
    }

    F32x8 S0, S1; S0.v = s0; S1.v = s1;
    float alpha[8];
#pragma unroll
    for (int r = 0; r < 8; ++r) {
      const float x0 = S0.f[r] * scale, x1 = S1.f[r] * scale;
      const float tmax = hmax16(fmaxf(x0, x1));
      const float mnew = fmaxf(mrun[r], tmax);
      const float al   = __expf(mrun[r] - mnew);
      const float p0   = __expf(x0 - mnew);
      const float p1   = __expf(x1 - mnew);
      lrun[r] = lrun[r] * al + hsum16(p0 + p1);
      mrun[r] = mnew; alpha[r] = al;
      S0.f[r] = p0; S1.f[r] = p1;
    }
#pragma unroll
    for (int r = 0; r < 8; ++r) {
      a0[r] *= alpha[r]; a1[r] *= alpha[r]; a2[r] *= alpha[r]; a3[r] *= alpha[r];
    }
    // C-layout -> row-major LDS (16x32 bf16), then read back as A fragment
    const int prow = (lane >> 4) << 3, pcol = lane & 15;
#pragma unroll
    for (int r = 0; r < 8; ++r) {
      p_lds[(prow + r) * 32 + pcol]      = f2bf(S0.f[r]);
      p_lds[(prow + r) * 32 + pcol + 16] = f2bf(S1.f[r]);
    }
    asm volatile("s_wait_dscnt 0" ::: "memory");
    const v16bf aP = load_a_lds(p_lds, lane);
    a0 = wmma_bf16(aP, bV0, a0);
    a1 = wmma_bf16(aP, bV1, a1);
    a2 = wmma_bf16(aP, bV2, a2);
    a3 = wmma_bf16(aP, bV3, a3);
    asm volatile("" ::: "memory");                    // order next LDS stores after reads
  }

  // epilogue: per-row dot with folded boundary weights, one float per token
  const int hh = bh & (Hv - 1);
  const int d  = lane & 15;
  const float w0 = weff[hh * HDv + d];
  const float w1 = weff[hh * HDv + 16 + d];
  const float w2 = weff[hh * HDv + 32 + d];
  const float w3 = weff[hh * HDv + 48 + d];
  F32x8 C0, C1, C2, C3; C0.v = a0; C1.v = a1; C2.v = a2; C3.v = a3;
#pragma unroll
  for (int r = 0; r < 8; ++r) {
    const float inv = 1.0f / fmaxf(lrun[r], 1e-20f);
    const float val = inv * (C0.f[r] * w0 + C1.f[r] * w1 + C2.f[r] * w2 + C3.f[r] * w3);
    const float rs  = hsum16(val);
    if ((lane & 15) == 0) {
      const int t = q0 + r + ((lane >> 4) << 3);
      part[(size_t)bh * Tv + t] = rs;
    }
  }
}

// ============ K4: combine heads -> boundary score (deterministic) =============
__global__ void k_bscore(const float* __restrict__ part, const float* __restrict__ weff,
                         float* __restrict__ bs) {
  const int i = blockIdx.x * blockDim.x + threadIdx.x;
  if (i >= Mv) return;
  const int b = i >> 11, t = i & (Tv - 1);
  float z = weff[Ev];                    // folded bias
#pragma unroll
  for (int h = 0; h < Hv; ++h) z += part[((size_t)(b * Hv + h)) * Tv + t];
  bs[i] = 1.0f / (1.0f + __expf(-z));
}

// ============ K5: per-batch cumsum -> patch ids ===============================
__global__ void k_pid(const float* __restrict__ bs, float* __restrict__ cum,
                      int* __restrict__ pid) {
  const int b = blockIdx.x, lane = threadIdx.x;
  float carry = 0.f;
  for (int base = 0; base < Tv; base += 32) {
    float v = bs[b * Tv + base + lane];
#pragma unroll
    for (int off = 1; off < 32; off <<= 1) {
      const float n = __shfl_up(v, off, 32);
      if (lane >= off) v += n;
    }
    v += carry;
    cum[b * Tv + base + lane] = v;
    carry = __shfl(v, 31, 32);
  }
  const float denom = fmaxf(carry, 1e-6f);
  for (int base = 0; base < Tv; base += 32) {
    const float c = cum[b * Tv + base + lane];
    int p = (int)((c / denom) * (float)Pv);
    pid[b * Tv + base + lane] = p < (Pv - 1) ? p : (Pv - 1);
  }
}

// ============ K6: mean-pool h into P patches + final projection ===============
__global__ void k_pool(const float* __restrict__ h, const int* __restrict__ pid,
                       const float* __restrict__ wpr, const float* __restrict__ bpr,
                       float* __restrict__ out) {
  __shared__ float s_pe[Pv * Ev];
  __shared__ int   s_cnt[Pv];
  const int b = blockIdx.x, e = threadIdx.x;       // 256 threads
  if (e < Pv) s_cnt[e] = 0;
  __syncthreads();
  for (int t = e; t < Tv; t += 256) atomicAdd(&s_cnt[pid[b * Tv + t]], 1);
  __syncthreads();
  float acc[Pv] = {0.f};
  for (int t = 0; t < Tv; ++t) {
    const int pp = pid[b * Tv + t];
    const float hv = h[((size_t)b * Tv + t) * Ev + e];
#pragma unroll
    for (int p = 0; p < Pv; ++p) acc[p] += (pp == p) ? hv : 0.f;
  }
#pragma unroll
  for (int p = 0; p < Pv; ++p)
    s_pe[p * Ev + e] = acc[p] / fmaxf((float)s_cnt[p], 1.0f);
  __syncthreads();
  float res[Pv] = {0.f};
  const float* wrow = wpr + (size_t)e * Ev;
  for (int kkk = 0; kkk < Ev; ++kkk) {
    const float w = wrow[kkk];
#pragma unroll
    for (int p = 0; p < Pv; ++p) res[p] += s_pe[p * Ev + kkk] * w;
  }
  const float bo = bpr[e];
#pragma unroll
  for (int p = 0; p < Pv; ++p)
    out[((size_t)b * Pv + p) * Ev + e] = res[p] + bo;
}

// ============ launcher ========================================================
extern "C" void kernel_launch(void* const* d_in, const int* in_sizes, int n_in,
                              void* d_out, int out_size, void* d_ws, size_t ws_size,
                              hipStream_t stream) {
  (void)in_sizes; (void)n_in; (void)out_size; (void)ws_size;
  const float* x    = (const float*)d_in[0];
  const float* w_in = (const float*)d_in[1];
  const float* b_in = (const float*)d_in[2];
  const float* w_q  = (const float*)d_in[3];
  const float* b_q  = (const float*)d_in[4];
  const float* w_k  = (const float*)d_in[5];
  const float* b_k  = (const float*)d_in[6];
  const float* w_v  = (const float*)d_in[7];
  const float* b_v  = (const float*)d_in[8];
  const float* w_o  = (const float*)d_in[9];
  const float* b_o  = (const float*)d_in[10];
  const float* w_bd = (const float*)d_in[11];
  const float* b_bd = (const float*)d_in[12];
  const float* w_pr = (const float*)d_in[13];
  const float* b_pr = (const float*)d_in[14];
  float* out = (float*)d_out;

  char* ws = (char*)d_ws;
  size_t off = 0;
  float*          h32  = (float*)         (ws + off); off += (size_t)Mv * Ev * 4;  // 16 MB
  unsigned short* h16  = (unsigned short*)(ws + off); off += (size_t)Mv * Ev * 2;  //  8 MB
  unsigned short* x16  = (unsigned short*)(ws + off); off += (size_t)Mv * DINv * 2;//  8 MB
  unsigned short* qb   = (unsigned short*)(ws + off); off += (size_t)Mv * Ev * 2;  //  8 MB
  unsigned short* kb   = (unsigned short*)(ws + off); off += (size_t)Mv * Ev * 2;  //  8 MB
  unsigned short* vT   = (unsigned short*)(ws + off); off += (size_t)Mv * Ev * 2;  //  8 MB
  unsigned short* win16= (unsigned short*)(ws + off); off += (size_t)Ev * DINv * 2;
  unsigned short* wq16 = (unsigned short*)(ws + off); off += (size_t)Ev * Ev * 2;
  unsigned short* wk16 = (unsigned short*)(ws + off); off += (size_t)Ev * Ev * 2;
  unsigned short* wv16 = (unsigned short*)(ws + off); off += (size_t)Ev * Ev * 2;
  float*          weff = (float*)         (ws + off); off += 4096;
  float*          part = (float*)         (ws + off); off += (size_t)Bv * Hv * Tv * 4;
  float*          bsv  = (float*)         (ws + off); off += (size_t)Mv * 4;
  float*          cum  = (float*)         (ws + off); off += (size_t)Mv * 4;
  int*            pid  = (int*)           (ws + off); off += (size_t)Mv * 4;

  dim3 wv32(32);
  // one-time precision conversion + linear-chain folding
  const int n4x = Mv * DINv / 4, n4w = Ev * Ev / 4;
  k_cvt <<<dim3((n4x + 255) / 256), dim3(256), 0, stream>>>(x, x16, n4x);
  k_cvt <<<dim3((n4w + 255) / 256), dim3(256), 0, stream>>>(w_in, win16, n4w);
  k_cvt <<<dim3((n4w + 255) / 256), dim3(256), 0, stream>>>(w_q, wq16, n4w);
  k_cvt <<<dim3((n4w + 255) / 256), dim3(256), 0, stream>>>(w_k, wk16, n4w);
  k_cvt <<<dim3((n4w + 255) / 256), dim3(256), 0, stream>>>(w_v, wv16, n4w);
  k_fold<<<dim3(1), dim3(256), 0, stream>>>(w_o, b_o, w_bd, b_bd, weff);
  // WMMA pipeline
  k_hproj<<<dim3(Mv / 32, Ev / 64), wv32, 0, stream>>>(x16, win16, b_in, h32, h16);
  k_qkv  <<<dim3(Mv / 32, Ev / 64, 3), wv32, 0, stream>>>(h16, wq16, wk16, wv16,
                                                          b_q, b_k, b_v, qb, kb, vT);
  k_attn <<<dim3(Bv * Hv, Tv / 16), wv32, 0, stream>>>(qb, kb, vT, weff, part);
  // scalar tail
  k_bscore<<<dim3(Mv / 256), dim3(256), 0, stream>>>(part, weff, bsv);
  k_pid   <<<dim3(Bv), wv32, 0, stream>>>(bsv, cum, pid);
  k_pool  <<<dim3(Bv), dim3(256), 0, stream>>>(h32, pid, w_pr, b_pr, out);
}
